// MixtureOfExperts_15728170238375
// MI455X (gfx1250) — matrix operand order repaired and checked by reading, not compiled, not run on hardware
//
#include <hip/hip_runtime.h>

// ---------------- problem constants (B=1) ----------------
constexpr int kS = 2048;   // tokens
constexpr int kH = 1024;   // hidden
constexpr int kI = 2048;   // intermediate
constexpr int kE = 8;      // experts

// ---------------- CDNA5 WMMA / TDM types ----------------
typedef __attribute__((ext_vector_type(16))) __bf16 v16bf;
typedef __attribute__((ext_vector_type(8)))  float  v8f;
typedef __attribute__((ext_vector_type(4)))  unsigned int u32x4;
typedef __attribute__((ext_vector_type(8)))  unsigned int u32x8;

union BF16Frag {
    uint4 q[2];   // two 16B chunks = 16 bf16 halves = 8 VGPRs
    v16bf v;
};

// fp32 -> bf16 with round-to-nearest-even (ignores NaN edge case)
__device__ __forceinline__ unsigned short f2bf(float f) {
    unsigned int u = __float_as_uint(f);
    u += 0x7FFFu + ((u >> 16) & 1u);
    return (unsigned short)(u >> 16);
}

// LDS byte offset of a __shared__ object (flat LDS aperture keeps the
// offset in addr[31:0] per CDNA5 ISA 10.2)
__device__ __forceinline__ unsigned lds_off(const void* p) {
    return (unsigned)(unsigned long long)p;
}

// ---------------- kernel 0: zero out, reset counts, x -> bf16 ----------------
__global__ void moe_init_kernel(const float* __restrict__ x,
                                float* __restrict__ out,
                                unsigned short* __restrict__ xb,
                                int* __restrict__ cnt) {
    int i = blockIdx.x * blockDim.x + threadIdx.x;
    if (i < kS * kH) {
        out[i] = 0.0f;
        xb[i]  = f2bf(x[i]);
    }
    if (i < kE) cnt[i] = 0;
}

// ---------------- kernel 1: router (one wave32 per token) ----------------
__global__ void moe_router_kernel(const float* __restrict__ x,
                                  const float* __restrict__ Wr,
                                  const float* __restrict__ br,
                                  int*   __restrict__ cnt,
                                  int*   __restrict__ ids,
                                  float* __restrict__ wts) {
    const int tid  = threadIdx.x;
    const int lane = tid & 31;          // wave32
    const int wid  = tid >> 5;
    const int t    = blockIdx.x * 8 + wid;

    float lg[kE];
#pragma unroll
    for (int e = 0; e < kE; ++e) lg[e] = 0.0f;

    // each lane strides H by 32; Wr row h is 8 contiguous floats (two float4)
    for (int h = lane; h < kH; h += 32) {
        float xv = x[(size_t)t * kH + h];
        const float4* wr = reinterpret_cast<const float4*>(Wr + (size_t)h * kE);
        float4 a = wr[0], b = wr[1];
        lg[0] += xv * a.x; lg[1] += xv * a.y; lg[2] += xv * a.z; lg[3] += xv * a.w;
        lg[4] += xv * b.x; lg[5] += xv * b.y; lg[6] += xv * b.z; lg[7] += xv * b.w;
    }
#pragma unroll
    for (int e = 0; e < kE; ++e) {
        float v = lg[e];
#pragma unroll
        for (int off = 16; off > 0; off >>= 1) v += __shfl_xor(v, off);
        lg[e] = v + br[e];
    }

    if (lane == 0) {
        int i1 = 0;
#pragma unroll
        for (int e = 1; e < kE; ++e) if (lg[e] > lg[i1]) i1 = e;
        int i2 = (i1 == 0) ? 1 : 0;
#pragma unroll
        for (int e = 0; e < kE; ++e) if (e != i1 && lg[e] > lg[i2]) i2 = e;
        // softmax over the two selected logits
        float ex = __expf(lg[i2] - lg[i1]);
        float w1 = 1.0f / (1.0f + ex);
        float w2 = ex / (1.0f + ex);
        int p1 = atomicAdd(&cnt[i1], 1);
        ids[i1 * kS + p1] = t;  wts[i1 * kS + p1] = w1;
        int p2 = atomicAdd(&cnt[i2], 1);
        ids[i2 * kS + p2] = t;  wts[i2 * kS + p2] = w2;
    }
}

// ---------------- kernel 2: fused expert FFN ----------------
// Block = 256 threads (8 wave32). Tile: (expert e, 64 gathered tokens, 128-wide
// slice of I). Phase 1: gate/up = x_tile @ W1/W3 slice via wmma bf16 with the
// token gather done by the Tensor Data Mover in gather mode (async, TENSORcnt);
// SiLU*up -> h tile (64x128 bf16) in LDS. Phase 2: partial down-proj
// y += h @ W2 slice (K=128), weighted scatter-add to out.
__global__ void __launch_bounds__(256)
moe_expert_kernel(const float* __restrict__ W1,
                  const float* __restrict__ W3,
                  const float* __restrict__ W2,
                  const int*   __restrict__ cnt,
                  const int*   __restrict__ ids,
                  const float* __restrict__ wts,
                  const unsigned short* __restrict__ xb,
                  float* __restrict__ out) {
    const int n0   = blockIdx.x * 128;   // I-slice base
    const int mt64 = blockIdx.y;         // 64-token tile index within expert list
    const int e    = blockIdx.z;
    const int ce   = cnt[e];
    if (mt64 * 64 >= ce) return;         // no tokens for this tile

    const int tid  = threadIdx.x;
    const int lane = tid & 31;           // wave32
    const int wid  = tid >> 5;           // 8 waves, each owns a 16-col sub-tile

    __shared__ __align__(16) unsigned short a_sm[64 * 32];     //  4 KB x tile (bf16)
    __shared__ __align__(16) unsigned short b1_sm[128 * 40];   // 10 KB W1 tile (transposed, padded)
    __shared__ __align__(16) unsigned short b3_sm[128 * 40];   // 10 KB W3 tile
    __shared__ __align__(16) unsigned short b2_sm[128 * 40];   // 10 KB W2 tile
    __shared__ __align__(16) unsigned short h_sm[64 * 128];    // 16 KB h tile (bf16)
    __shared__ int   tok_sm[64];
    __shared__ float w_sm[64];

    if (tid < 64) {
        int lt = mt64 * 64 + tid;
        bool valid = lt < ce;
        tok_sm[tid] = valid ? ids[e * kS + lt] : 0;
        w_sm[tid]   = valid ? wts[e * kS + lt] : 0.0f;
    }
    __syncthreads();

    // per-lane fragment geometry (ISA 7.12.2, wave32)
    const unsigned nloc  = lane & 15;
    const unsigned bko   = (lane < 16) ? 0u : 16u;  // B: K 0..15 | 16..31 (contiguous)
    const unsigned ako   = (lane < 16) ? 0u : 16u;  // A: K 0..15 | 16..31 (contiguous)
    const unsigned mhalf = (lane < 16) ? 0u : 8u;   // C: rows r | 8+r
    const unsigned bn    = wid * 16 + nloc;         // this wave's column in 128-wide tile

    // ---- TDM gather setup: this wave's A sub-tile (16 token rows x 32 bf16).
    // Waves 0-3 own M-tiles 0-3; waves 4-7 issue duplicates (each wave must
    // track its own TENSORcnt). Indices are 16-bit, packed into D# groups 2/3,
    // made wave-uniform so they live in SGPRs.
    const int mt_tdm = wid & 3;
    unsigned gidx[8];
#pragma unroll
    for (int j = 0; j < 8; ++j) {
        int l0 = mt64 * 64 + mt_tdm * 16 + 2 * j;
        int v0 = (l0     < ce) ? ids[e * kS + l0]     : 0;
        int v1 = (l0 + 1 < ce) ? ids[e * kS + l0 + 1] : 0;
        gidx[j] = (unsigned)__builtin_amdgcn_readfirstlane((v0 & 0xFFFF) | (v1 << 16));
    }
    const u32x4 tdm_g2 = {gidx[0], gidx[1], gidx[2], gidx[3]};
    const u32x4 tdm_g3 = {gidx[4], gidx[5], gidx[6], gidx[7]};
    const unsigned lds_a = lds_off(a_sm) + (unsigned)(mt_tdm * 16 * 64); // bytes
    // D# group1: wg_mask=0, data_size=2B(code 1), tensor_dim0=32, tensor_dim1=S,
    // tile_dim0=32, tile_dim1=16 valid indices, tensor_dim0_stride=H
    const u32x8 tdm_g1 = {
        0x00010000u,                 // [17:16] data_size=1 -> 2 bytes
        (32u << 16),                 // tensor_dim0[15:0] in bits [63:48]
        ((unsigned)kS << 16),        // tensor_dim1[15:0] in bits [95:80]
        (32u << 16),                 // tile_dim0 in bits [127:112]
        16u,                         // tile_dim1 = #valid gather indices
        (unsigned)kH,                // tensor_dim0_stride (elements)
        0u, 0u
    };

    const v8f vzero = {0.f, 0.f, 0.f, 0.f, 0.f, 0.f, 0.f, 0.f};
    v8f accg[4], accu[4];
#pragma unroll
    for (int m = 0; m < 4; ++m) { accg[m] = vzero; accu[m] = vzero; }

    // ---- Phase 1: gate/up GEMMs, K over H in chunks of 32 ----
    for (int kc = 0; kc < kH / 32; ++kc) {
        __syncthreads();    // previous iteration's LDS reads retired (dscnt+barrier)

        {   // async token-row gather: global bf16 x rows -> LDS A tile (TDM)
            unsigned long long ga =
                (unsigned long long)xb + (unsigned long long)kc * 64ull; // col offset
            u32x4 g0 = { 0x80000001u,            // count=1, gather_mode=1, 16-bit idx
                         lds_a,
                         (unsigned)ga,
                         ((unsigned)(ga >> 32) & 0x01FFFFFFu) | 0x80000000u }; // type=2
            asm volatile("tensor_load_to_lds %0, %1, %2, %3"
                         :: "s"(g0), "s"(tdm_g1), "s"(tdm_g2), "s"(tdm_g3)
                         : "memory");
        }

        if (kc + 1 < kH / 32) {   // gfx1250 global_prefetch_b8 of next W chunk
            size_t pb = ((size_t)e * kH + (size_t)(kc + 1) * 32) * kI + n0;
            __builtin_prefetch(W1 + pb + (size_t)(tid >> 3) * kI + (tid & 7) * 16, 0, 0);
            __builtin_prefetch(W3 + pb + (size_t)(tid >> 3) * kI + (tid & 7) * 16, 0, 0);
        }
        // stage W1/W3 tiles (f32 -> bf16, transposed, padded stride 40 halves);
        // overlaps with the in-flight tensor DMA above
        for (int idx = tid; idx < 32 * 128; idx += 256) {
            int k = idx >> 7, n = idx & 127;
            size_t off = ((size_t)e * kH + kc * 32 + k) * kI + n0 + n;
            b1_sm[n * 40 + k] = f2bf(W1[off]);
            b3_sm[n * 40 + k] = f2bf(W3[off]);
        }
        __builtin_amdgcn_s_wait_tensorcnt(0);   // A tile landed in LDS
        __syncthreads();                        // publish A + B tiles to all waves

        BF16Frag fb1, fb3;
        fb1.q[0] = *reinterpret_cast<const uint4*>(b1_sm + bn * 40 + bko);
        fb1.q[1] = *reinterpret_cast<const uint4*>(b1_sm + bn * 40 + bko + 8);
        fb3.q[0] = *reinterpret_cast<const uint4*>(b3_sm + bn * 40 + bko);
        fb3.q[1] = *reinterpret_cast<const uint4*>(b3_sm + bn * 40 + bko + 8);
#pragma unroll
        for (int m = 0; m < 4; ++m) {
            BF16Frag fa;
            unsigned arow = m * 16 + nloc;
            fa.q[0] = *reinterpret_cast<const uint4*>(a_sm + arow * 32 + ako);
            fa.q[1] = *reinterpret_cast<const uint4*>(a_sm + arow * 32 + ako + 8);
            accg[m] = __builtin_amdgcn_wmma_f32_16x16x32_bf16(
                false, fa.v, false, fb1.v, (short)0, accg[m], false, false);
            accu[m] = __builtin_amdgcn_wmma_f32_16x16x32_bf16(
                false, fa.v, false, fb3.v, (short)0, accu[m], false, false);
        }
    }

    // ---- SwiGLU in registers, park h tile in LDS as bf16 ----
    __syncthreads();
#pragma unroll
    for (int m = 0; m < 4; ++m) {
#pragma unroll
        for (int r = 0; r < 8; ++r) {
            float g = accg[m][r], u = accu[m][r];
            float hv = (g / (1.0f + __expf(-g))) * u;   // silu(g) * up
            unsigned row = m * 16 + mhalf + r;
            unsigned col = wid * 16 + nloc;
            h_sm[row * 128 + col] = f2bf(hv);
        }
    }
    __syncthreads();

    // ---- Phase 2: down-proj partial (K = this 128-wide I slice) ----
    for (int ho = 0; ho < kH; ho += 128) {
        v8f acc[4];
#pragma unroll
        for (int m = 0; m < 4; ++m) acc[m] = vzero;
        for (int kc2 = 0; kc2 < 4; ++kc2) {
            __syncthreads();
            for (int idx = tid; idx < 32 * 128; idx += 256) {
                int k = idx >> 7, n = idx & 127;
                size_t off = ((size_t)e * kI + n0 + kc2 * 32 + k) * kH + ho + n;
                b2_sm[n * 40 + k] = f2bf(W2[off]);
            }
            __syncthreads();
            BF16Frag fb2;
            fb2.q[0] = *reinterpret_cast<const uint4*>(b2_sm + bn * 40 + bko);
            fb2.q[1] = *reinterpret_cast<const uint4*>(b2_sm + bn * 40 + bko + 8);
#pragma unroll
            for (int m = 0; m < 4; ++m) {
                BF16Frag fa;
                unsigned arow = m * 16 + nloc;
                unsigned base = arow * 128 + kc2 * 32 + ako;
                fa.q[0] = *reinterpret_cast<const uint4*>(h_sm + base);
                fa.q[1] = *reinterpret_cast<const uint4*>(h_sm + base + 8);
                acc[m] = __builtin_amdgcn_wmma_f32_16x16x32_bf16(
                    false, fa.v, false, fb2.v, (short)0, acc[m], false, false);
            }
        }
        // weighted scatter-add (global_atomic_add_f32); padded rows have w=0
        const int col = ho + wid * 16 + nloc;
#pragma unroll
        for (int m = 0; m < 4; ++m) {
#pragma unroll
            for (int r = 0; r < 8; ++r) {
                int lt = m * 16 + mhalf + r;
                float wgt = w_sm[lt];
                int t = tok_sm[lt];
                atomicAdd(out + (size_t)t * kH + col, wgt * acc[m][r]);
            }
        }
    }
}

// ---------------- host launcher ----------------
extern "C" void kernel_launch(void* const* d_in, const int* in_sizes, int n_in,
                              void* d_out, int out_size, void* d_ws, size_t ws_size,
                              hipStream_t stream) {
    (void)in_sizes; (void)n_in; (void)out_size; (void)ws_size;
    const float* x  = (const float*)d_in[0];
    const float* Wr = (const float*)d_in[1];
    const float* br = (const float*)d_in[2];
    const float* W1 = (const float*)d_in[3];
    const float* W3 = (const float*)d_in[4];
    const float* W2 = (const float*)d_in[5];
    float* out = (float*)d_out;

    // workspace layout (re-initialized every call; ~4.3 MB total)
    char* ws = (char*)d_ws;
    int*   cnt = (int*)ws;                                        // kE counts
    int*   ids = (int*)(ws + 1024);                               // kE*kS token ids
    float* wts = (float*)(ws + 1024 + kE * kS * 4);               // kE*kS weights
    unsigned short* xb = (unsigned short*)(ws + 1024 + 2 * kE * kS * 4); // bf16 x

    moe_init_kernel<<<(kS * kH + 255) / 256, 256, 0, stream>>>(x, out, xb, cnt);
    moe_router_kernel<<<kS / 8, 256, 0, stream>>>(x, Wr, br, cnt, ids, wts);
    dim3 grid(kI / 128, kS / 64, kE);   // (I-slices, token tiles, experts)
    moe_expert_kernel<<<grid, 256, 0, stream>>>(W1, W3, W2, cnt, ids, wts, xb, out);
}
